// FeatureLevelAttention_56564719288687
// MI455X (gfx1250) — matrix operand order repaired and checked by reading, not compiled, not run on hardware
//
#include <hip/hip_runtime.h>

typedef __attribute__((ext_vector_type(16))) _Float16 v16h;
typedef __attribute__((ext_vector_type(8)))  _Float16 v8h;
typedef __attribute__((ext_vector_type(8)))  float    v8f;
typedef __attribute__((ext_vector_type(4)))  unsigned int v4u;
typedef __attribute__((ext_vector_type(8)))  int v8i;
typedef __attribute__((ext_vector_type(4)))  int v4i;

#define BATCH 8
#define HH    224
#define WW    224
#define HW    50176
#define OC    32

#if __has_builtin(__builtin_amdgcn_tensor_load_to_lds) && __has_builtin(__builtin_amdgcn_s_wait_tensorcnt)
#define HAVE_TDM 1
#endif

// 16-bit A-matrix 16x32 layout: lane L holds row M=L&15; element e covers two
// runs of 8 consecutive K: K = 8*(L>=16) + (e<8 ? e : e+8)  (+16 for e>=8 run).
__device__ __forceinline__ v8f wmma_f16(v16h a, v16h b, v8f c) {
    return __builtin_amdgcn_wmma_f32_16x16x32_f16(false, a, false, b, (short)0, c, false, false);
}

#ifdef HAVE_TDM
// Build a Tensor-DMA D# for a 2D f16 tile (row-major, tile_h rows of tile_w
// elems, rows strided by row_stride elems) and issue TENSOR_LOAD_TO_LDS.
// D# packing per CDNA5 ISA.
__device__ __forceinline__ void tdm_load_2d(const void* gptr, unsigned lds_byte_off,
                                            unsigned tile_w, unsigned tile_h,
                                            unsigned tensor_w, unsigned tensor_h,
                                            unsigned row_stride) {
    unsigned long long ga = (unsigned long long)(size_t)gptr;
    v4u g0; v8i g1; v4i z4 = {0, 0, 0, 0};
    g0[0] = 1u;                                             // count=1, user descriptor
    g0[1] = lds_byte_off;                                   // lds_addr (bytes)
    g0[2] = (unsigned)ga;                                   // global_addr[31:0] -> bits 95:64
    g0[3] = ((unsigned)(ga >> 32) & 0x01ffffffu) | 0x80000000u; // addr[56:32] | type=2
    g1[0] = (int)(1u << 16);                                // wg_mask=0, data_size=1 (2B)
    g1[1] = (int)((tensor_w & 0xffffu) << 16);              // tensor_dim0[15:0] @ bits63:48
    g1[2] = (int)(((tensor_w >> 16) & 0xffffu) | ((tensor_h & 0xffffu) << 16));
    g1[3] = (int)(((tensor_h >> 16) & 0xffffu) | ((tile_w & 0xffffu) << 16));   // tile_dim0
    g1[4] = (int)(tile_h & 0xffffu);                        // tile_dim1 (tile_dim2=0)
    g1[5] = (int)row_stride;                                // tensor_dim0_stride[31:0]
    g1[6] = 0;
    g1[7] = 0;
#if defined(__clang_major__) && (__clang_major__ >= 23)
    v8i z8 = {0, 0, 0, 0, 0, 0, 0, 0};
    __builtin_amdgcn_tensor_load_to_lds(g0, g1, z4, z4, z8, 0);
#else
    __builtin_amdgcn_tensor_load_to_lds(g0, g1, z4, z4, 0);
#endif
}
#endif

// ---------------------------------------------------------------------------
// Kernel 1: x1 = conv3x3(x, in_w) + in_b  (f32 + f16 copies)
// ---------------------------------------------------------------------------
__global__ void k_in_conv(const float* __restrict__ x, const float* __restrict__ w,
                          const float* __restrict__ b, float* __restrict__ x1f,
                          _Float16* __restrict__ x1h) {
    int idx = blockIdx.x * 256 + threadIdx.x;
    if (idx >= BATCH * OC * HW) return;
    int px = idx % WW, py = (idx / WW) % HH, oc = (idx / HW) % OC, bb = idx / (OC * HW);
    float acc = b[oc];
    for (int c = 0; c < 3; ++c)
        for (int ky = 0; ky < 3; ++ky) {
            int yy = py + ky - 1; if (yy < 0 || yy >= HH) continue;
            for (int kx = 0; kx < 3; ++kx) {
                int xx = px + kx - 1; if (xx < 0 || xx >= WW) continue;
                acc += x[((bb * 3 + c) * HH + yy) * WW + xx] * w[((oc * 3 + c) * 3 + ky) * 3 + kx];
            }
        }
    x1f[idx] = acc;
    x1h[idx] = (_Float16)acc;
}

// f32 -> f16 weight pre-conversion (qkv weights, 96*288 elems)
__global__ void k_prep(const float* __restrict__ w, _Float16* __restrict__ wh, int n) {
    int i = blockIdx.x * 256 + threadIdx.x;
    if (i < n) wh[i] = (_Float16)w[i];
}

// ---------------------------------------------------------------------------
// Kernel 2: qkv = conv3x3(x1, qkv_w) as implicit GEMM (M=16 px, N=96 ch, K=288)
// ---------------------------------------------------------------------------
__global__ void __launch_bounds__(128) k_qkv_wmma(const _Float16* __restrict__ x1h,
                                                  const _Float16* __restrict__ qwh,
                                                  _Float16* __restrict__ qkvh) {
    __shared__ _Float16 patch[3 * 18 * 32];        // rows y-1..y+1, cols x0-1..x0+16, 32 ch
    __shared__ unsigned short ctab[288];           // K -> patch offset (minus m*32 term)
    int bi = blockIdx.x;
    int xt = bi % 14, py = (bi / 14) % HH, bb = bi / (14 * HH);
    int x0 = xt * 16;
    int tid = threadIdx.x;
    for (int i = tid; i < 288; i += 128) {
        int c = i / 9, tap = i % 9;
        ctab[i] = (unsigned short)(((tap / 3) * 18 + (tap % 3)) * 32 + c);
    }
    for (int i = tid; i < 3 * 18 * 32; i += 128) {
        int c = i & 31, rest = i >> 5, col = rest % 18, ry = rest / 18;
        int yy = py + ry - 1, xx = x0 + col - 1;
        _Float16 v = (_Float16)0.f;
        if (yy >= 0 && yy < HH && xx >= 0 && xx < WW)
            v = x1h[(((size_t)bb * OC + c) * HH + yy) * WW + xx];
        patch[i] = v;
    }
    __syncthreads();
    int lane = tid & 31, wave = tid >> 5;
    int m = lane & 15, hi = lane >> 4;
    for (int nt = wave; nt < 6; nt += 4) {
        v8f acc = {};
#pragma unroll
        for (int t = 0; t < 9; ++t) {
            v16h a, bf;
#pragma unroll
            for (int e = 0; e < 16; ++e) {
                int K = 32 * t + (hi << 3) + (e < 8 ? e : e + 8);
                a[e] = patch[ctab[K] + (m << 5)];
            }
            int o = nt * 16 + m;
            const v8h* wp = (const v8h*)(qwh + (size_t)o * 288 + 32 * t + (hi << 3));
            v8h w0 = wp[0], w1 = wp[2];
#pragma unroll
            for (int e = 0; e < 8; ++e) { bf[e] = w0[e]; bf[8 + e] = w1[e]; }
            acc = wmma_f16(a, bf, acc);
        }
#pragma unroll
        for (int r = 0; r < 8; ++r) {
            int mm = r + (hi << 3);
            int o  = nt * 16 + m;
            qkvh[(((size_t)bb * 96 + o) * HH + py) * WW + x0 + mm] = (_Float16)acc[r];
        }
    }
}

// ---------------------------------------------------------------------------
// Kernel 3: fused attention per (slab, 16-row tile); k/v chunks staged to LDS
// by the Tensor Data Mover (double-buffered, s_wait_tensorcnt) when available.
// ---------------------------------------------------------------------------
__global__ void __launch_bounds__(128) k_attn(const _Float16* __restrict__ qkvh,
                                              const float* __restrict__ gumbel,
                                              float* __restrict__ att_out,
                                              float* __restrict__ Aout) {
    __shared__ _Float16 chunk[2][224 * 32];        // TDM-staged k/v chunk (14 KB each)
    __shared__ float    lg[16 * 224];
    __shared__ _Float16 atth[16 * 224];
    __shared__ float    pmax[16][8];
    __shared__ int      pidx[16][8];
    __shared__ float    psum[16][8];

    int it = blockIdx.x;                 // i-tile 0..13
    int slab = blockIdx.y;               // b*32 + n*4 + d
    int b = slab >> 5, ch = slab & 31, n = ch >> 2, d = ch & 3;
    const _Float16* qh = qkvh + (((size_t)b * 96 + n * 12 + d) * HW);
    const _Float16* kh = qkvh + (((size_t)b * 96 + n * 12 + 4 + d) * HW);
    const _Float16* vh = qkvh + (((size_t)b * 96 + n * 12 + 8 + d) * HW);
    int tid = threadIdx.x, lane = tid & 31, wave = tid >> 5;
    int m = lane & 15, hi = lane >> 4;
    int i0 = it * 16;

    // ---- Phase 1: logits = q k^T * 0.5 ----
    v16h qa[7];
#pragma unroll
    for (int t = 0; t < 7; ++t) {
        v8h a0 = *(const v8h*)(qh + (size_t)(i0 + m) * WW + 32 * t + 8 * hi);
        v8h a1 = *(const v8h*)(qh + (size_t)(i0 + m) * WW + 32 * t + 16 + 8 * hi);
#pragma unroll
        for (int e = 0; e < 8; ++e) { qa[t][e] = a0[e]; qa[t][8 + e] = a1[e]; }
    }
    v8f accj[4];
#pragma unroll
    for (int q = 0; q < 4; ++q) accj[q] = (v8f){0, 0, 0, 0, 0, 0, 0, 0};
#ifdef HAVE_TDM
    if (wave == 0)   // k chunk 0: cols 0..31 of all 224 rows
        tdm_load_2d(kh, (unsigned)(size_t)(void*)chunk[0], 32, 224, 224, 224, 224);
#endif
#pragma unroll
    for (int t = 0; t < 7; ++t) {
#ifdef HAVE_TDM
        if (wave == 0) {
            if (t < 6) {
                tdm_load_2d(kh + 32 * (t + 1), (unsigned)(size_t)(void*)chunk[(t + 1) & 1],
                            32, 224, 224, 224, 224);
                __builtin_amdgcn_s_wait_tensorcnt(1);
            } else {
                __builtin_amdgcn_s_wait_tensorcnt(0);
            }
        }
        const _Float16* cb = chunk[t & 1];
#else
        for (int g = tid; g < 896; g += 128) {         // cooperative fallback
            int j = g >> 2, c8 = (g & 3) << 3;
            *(v8h*)(chunk[0] + j * 32 + c8) = *(const v8h*)(kh + (size_t)j * WW + 32 * t + c8);
        }
        const _Float16* cb = chunk[0];
#endif
        __syncthreads();
#pragma unroll
        for (int q = 0; q < 4; ++q) {
            int jt = wave + 4 * q;
            if (jt < 14) {
                int j = jt * 16 + m;                   // B[K=w][N=j]=k[j][w]: contiguous LDS
                const v8h* bp = (const v8h*)(cb + j * 32 + (hi << 3));
                v8h b0 = bp[0], b1 = bp[2];
                v16h bf;
#pragma unroll
                for (int e = 0; e < 8; ++e) { bf[e] = b0[e]; bf[8 + e] = b1[e]; }
                accj[q] = wmma_f16(qa[t], bf, accj[q]);
            }
        }
        __syncthreads();
    }
    const float scale = 0.5f;                          // HD^-0.5
#pragma unroll
    for (int q = 0; q < 4; ++q) {
        int jt = wave + 4 * q;
        if (jt < 14)
#pragma unroll
            for (int r = 0; r < 8; ++r)
                lg[(r + (hi << 3)) * 224 + jt * 16 + m] = accj[q][r] * scale;
    }
    __syncthreads();

    // ---- Phase 2: +gumbel, softmax, argmax one-hot, straight-through ----
    {
        int row = tid >> 3, sub = tid & 7;             // j = c*8+sub -> coalesced
        const float* grow = gumbel + (size_t)slab * HW + (size_t)(i0 + row) * 224;
        float mx = -3.4e38f; int am = 0;
        for (int c = 0; c < 28; ++c) {
            int j = c * 8 + sub;
            float v = lg[row * 224 + j] + grow[j];
            lg[row * 224 + j] = v;
            if (v > mx) { mx = v; am = j; }
        }
        pmax[row][sub] = mx; pidx[row][sub] = am;
        __syncthreads();
        float rmx = pmax[row][0]; int ram = pidx[row][0];
        for (int s = 1; s < 8; ++s)
            if (pmax[row][s] > rmx) { rmx = pmax[row][s]; ram = pidx[row][s]; }
        float ps = 0.f;
        for (int c = 0; c < 28; ++c) {
            int j = c * 8 + sub;
            float e = __expf(lg[row * 224 + j] - rmx);
            lg[row * 224 + j] = e;
            ps += e;
        }
        psum[row][sub] = ps;
        __syncthreads();
        float tot = 0.f;
        for (int s = 0; s < 8; ++s) tot += psum[row][s];
        float inv = 1.f / tot;
        float* arow = att_out + (size_t)slab * HW + (size_t)(i0 + row) * 224;
        for (int c = 0; c < 28; ++c) {
            int j = c * 8 + sub;
            float ys = lg[row * 224 + j] * inv;
            float yh = (j == ram) ? 1.f : 0.f;
            float a = (yh - ys) + ys;                  // straight-through (fwd == hard)
            arow[j] = a;
            atth[row * 224 + j] = (_Float16)a;
        }
    }
    __syncthreads();

    // ---- Phase 3: A = att * v ----
#ifdef HAVE_TDM
    if (wave == 0)   // v chunk 0: rows 0..31, full 224 cols
        tdm_load_2d(vh, (unsigned)(size_t)(void*)chunk[0], 224, 32, 224, 224, 224);
#endif
    v16h aa[7];
#pragma unroll
    for (int t = 0; t < 7; ++t) {                      // contiguous LDS runs of 8 halves
        v8h a0 = *(const v8h*)(atth + m * 224 + 32 * t + 8 * hi);
        v8h a1 = *(const v8h*)(atth + m * 224 + 32 * t + 16 + 8 * hi);
#pragma unroll
        for (int e = 0; e < 8; ++e) { aa[t][e] = a0[e]; aa[t][8 + e] = a1[e]; }
    }
    v8f accw[4];
#pragma unroll
    for (int q = 0; q < 4; ++q) accw[q] = (v8f){0, 0, 0, 0, 0, 0, 0, 0};
#pragma unroll
    for (int t = 0; t < 7; ++t) {
#ifdef HAVE_TDM
        if (wave == 0) {
            if (t < 6) {
                tdm_load_2d(vh + (size_t)32 * (t + 1) * WW,
                            (unsigned)(size_t)(void*)chunk[(t + 1) & 1], 224, 32, 224, 224, 224);
                __builtin_amdgcn_s_wait_tensorcnt(1);
            } else {
                __builtin_amdgcn_s_wait_tensorcnt(0);
            }
        }
        const _Float16* cb = chunk[t & 1];
#else
        for (int g = tid; g < 896; g += 128) {
            int jl = g / 28, c8 = (g % 28) << 3;
            *(v8h*)(chunk[0] + jl * 224 + c8) =
                *(const v8h*)(vh + (size_t)(32 * t + jl) * WW + c8);
        }
        const _Float16* cb = chunk[0];
#endif
        __syncthreads();
#pragma unroll
        for (int q = 0; q < 4; ++q) {
            int wt = wave + 4 * q;
            if (wt < 14) {
                int w = wt * 16 + m;                   // B[K=j][N=w]=v[j][w]: LDS gather
                v16h bf;
#pragma unroll
                for (int e = 0; e < 16; ++e)
                    bf[e] = cb[((hi << 3) + (e < 8 ? e : e + 8)) * 224 + w];
                accw[q] = wmma_f16(aa[t], bf, accw[q]);
            }
        }
        __syncthreads();
    }
#pragma unroll
    for (int q = 0; q < 4; ++q) {
        int wt = wave + 4 * q;
        if (wt < 14)
#pragma unroll
            for (int r = 0; r < 8; ++r)
                Aout[(size_t)slab * HW + (size_t)(i0 + r + (hi << 3)) * 224 + wt * 16 + m] = accw[q][r];
    }
}

// ---------------------------------------------------------------------------
// Kernel 4: compose 1x1 convs: Wc = upd_w @ proj_w ; bc = upd_w@proj_b + upd_b
// ---------------------------------------------------------------------------
__global__ void k_wcomb(const float* __restrict__ pw, const float* __restrict__ pb,
                        const float* __restrict__ uw, const float* __restrict__ ub,
                        float* __restrict__ wc, float* __restrict__ bc) {
    int tid = threadIdx.x;
    for (int idx = tid; idx < 1024; idx += 256) {
        int o = idx >> 5, i = idx & 31;
        float s = 0.f;
        for (int mm = 0; mm < 32; ++mm) s += uw[o * 32 + mm] * pw[mm * 32 + i];
        wc[idx] = s;
    }
    if (tid < 32) {
        float s = ub[tid];
        for (int mm = 0; mm < 32; ++mm) s += uw[tid * 32 + mm] * pb[mm];
        bc[tid] = s;
    }
}

// ---------------------------------------------------------------------------
// Kernel 5: A2 = Wc * A + bc per pixel (no reductions -> deterministic)
// ---------------------------------------------------------------------------
__global__ void k_mix(const float* __restrict__ A, const float* __restrict__ wc,
                      const float* __restrict__ bc, float* __restrict__ A2) {
    __shared__ float swc[1024], sbc[32];
    int tid = threadIdx.x;
    for (int i = tid; i < 1024; i += 256) swc[i] = wc[i];
    if (tid < 32) sbc[tid] = bc[tid];
    __syncthreads();
    size_t p = (size_t)blockIdx.x * 256 + tid;          // pixel over B*H*W
    int bb = (int)(p / HW), pix = (int)(p % HW);
    float av[32];
    for (int c = 0; c < 32; ++c) av[c] = A[((size_t)bb * 32 + c) * HW + pix];
    for (int o = 0; o < 32; ++o) {
        float s = sbc[o];
        for (int c = 0; c < 32; ++c) s += swc[o * 32 + c] * av[c];
        A2[((size_t)bb * 32 + o) * HW + pix] = s;
    }
}

// ---------------------------------------------------------------------------
// Kernel 6: per-channel sum/sumsq, fixed-order (deterministic, no atomics)
// one block per channel
// ---------------------------------------------------------------------------
__global__ void k_stats(const float* __restrict__ A2, float* __restrict__ gsum) {
    __shared__ float ssum[256], ssq[256];
    int chn = blockIdx.x, tid = threadIdx.x;
    float s = 0.f, q = 0.f;
    for (int b = 0; b < BATCH; ++b) {
        const float* p = A2 + ((size_t)b * 32 + chn) * HW;
        for (int i = tid; i < HW; i += 256) { float v = p[i]; s += v; q += v * v; }
    }
    ssum[tid] = s; ssq[tid] = q;
    __syncthreads();
    for (int off = 128; off > 0; off >>= 1) {
        if (tid < off) { ssum[tid] += ssum[tid + off]; ssq[tid] += ssq[tid + off]; }
        __syncthreads();
    }
    if (tid == 0) { gsum[chn] = ssum[0]; gsum[32 + chn] = ssq[0]; }
}

// ---------------------------------------------------------------------------
// Kernel 7: batchnorm + SiLU + residual
// ---------------------------------------------------------------------------
__global__ void k_final(const float* __restrict__ A2, const float* __restrict__ x1f,
                        const float* __restrict__ gsum, const float* __restrict__ gamma,
                        const float* __restrict__ beta, float* __restrict__ out) {
    size_t idx = (size_t)blockIdx.x * 256 + threadIdx.x;
    if (idx >= (size_t)BATCH * OC * HW) return;
    int chn = (int)((idx / HW) % OC);
    const float N = (float)(BATCH * HW);
    float mean = gsum[chn] / N;
    float var  = gsum[32 + chn] / N - mean * mean;
    float a = (A2[idx] - mean) * rsqrtf(var + 1e-5f);
    a = a * gamma[chn] + beta[chn];
    float s = a / (1.f + __expf(-a));
    out[idx] = s + x1f[idx];
}

// ---------------------------------------------------------------------------
extern "C" void kernel_launch(void* const* d_in, const int* in_sizes, int n_in,
                              void* d_out, int out_size, void* d_ws, size_t ws_size,
                              hipStream_t stream) {
    (void)in_sizes; (void)n_in; (void)out_size; (void)ws_size;
    const float* x      = (const float*)d_in[0];
    const float* gumbel = (const float*)d_in[1];
    const float* in_w   = (const float*)d_in[2];
    const float* in_b   = (const float*)d_in[3];
    const float* qkv_w  = (const float*)d_in[4];
    const float* proj_w = (const float*)d_in[5];
    const float* proj_b = (const float*)d_in[6];
    const float* upd_w  = (const float*)d_in[7];
    const float* upd_b  = (const float*)d_in[8];
    const float* gamma  = (const float*)d_in[9];
    const float* beta   = (const float*)d_in[10];

    char* ws = (char*)d_ws;
    size_t off = 0;
    float*    x1f  = (float*)(ws + off);    off += (size_t)BATCH * OC * HW * 4;
    _Float16* x1h  = (_Float16*)(ws + off); off += (size_t)BATCH * OC * HW * 2;
    _Float16* qkvh = (_Float16*)(ws + off); off += (size_t)BATCH * 96 * HW * 2;
    float*    Aws  = (float*)(ws + off);    off += (size_t)BATCH * OC * HW * 4;
    float*    A2   = (float*)(ws + off);    off += (size_t)BATCH * OC * HW * 4;
    float*    wc   = (float*)(ws + off);    off += 4096;
    float*    bc   = (float*)(ws + off);    off += 256;
    float*    gsum = (float*)(ws + off);    off += 256;
    _Float16* qwh  = (_Float16*)(ws + off); off += 96 * 288 * 2;

    float* out_main = (float*)d_out;
    float* att_out  = out_main + (size_t)BATCH * OC * HW;   // (out, att) concatenated

    k_in_conv <<<(BATCH * OC * HW) / 256, 256, 0, stream>>>(x, in_w, in_b, x1f, x1h);
    k_prep    <<<108, 256, 0, stream>>>(qkv_w, qwh, 96 * 288);
    k_qkv_wmma<<<BATCH * HH * (WW / 16), 128, 0, stream>>>(x1h, qwh, qkvh);
    k_attn    <<<dim3(14, 256), 128, 0, stream>>>(qkvh, gumbel, att_out, Aws);
    k_wcomb   <<<1, 256, 0, stream>>>(proj_w, proj_b, upd_w, upd_b, wc, bc);
    k_mix     <<<(BATCH * HW) / 256, 256, 0, stream>>>(Aws, wc, bc, A2);
    k_stats   <<<32, 256, 0, stream>>>(A2, gsum);
    k_final   <<<(BATCH * OC * HW) / 256, 256, 0, stream>>>(A2, x1f, gsum, gamma, beta, out_main);
}
